// MPNN_54803782697380
// MI455X (gfx1250) — compile-verified, matrix operand order checked by reading, not run on hardware
//
#include <hip/hip_runtime.h>
#include <hip/hip_bf16.h>

typedef _Float16 h16;
typedef __attribute__((ext_vector_type(16))) _Float16 v16h;
typedef __attribute__((ext_vector_type(8)))  _Float16 v8h;
typedef __attribute__((ext_vector_type(8)))  float    v8f;

#define NN 48
#define EE 2256
#define HH 256
#define BB 32
#define PRED 8

// ---------------------------------------------------------------------------
// Fragment loader: 16x32 f16 tile (row stride `ld` elements) -> WMMA A layout.
// Per CDNA5 ISA 7.12.2 (16-bit A-matrix 16x32): lane L<16 holds row L with
// K = 0..7 and 16..23; lane L+16 holds row L with K = 8..15 and 24..31.
// B fragments use the same layout loaded from B-transposed (NxK row-major).
// ---------------------------------------------------------------------------
__device__ __forceinline__ v16h ld_frag(const h16* __restrict__ base, int ld) {
  const int lane = threadIdx.x & 31;
  const int r    = lane & 15;
  const int koff = (lane >> 4) << 3;   // 0 or 8
  const h16* p = base + r * ld + koff;
  v8h lo = *(const v8h*)(p);
  v8h hi = *(const v8h*)(p + 16);
  v16h f;
#pragma unroll
  for (int i = 0; i < 8; ++i) { f[i] = lo[i]; f[i + 8] = hi[i]; }
  return f;
}

__device__ __forceinline__ v8f wmma_f16(v16h a, v16h b, v8f c) {
  return __builtin_amdgcn_wmma_f32_16x16x32_f16(false, a, false, b, (short)0, c,
                                                false, false);
}

// ---------------------------------------------------------------------------
// Prep: copy initial state (layout-identical) and build f16 transposed weights
// dst[n*Kpad + k] = (k < Kin) ? src[k*Nout + n] : 0
// ---------------------------------------------------------------------------
__global__ void copy_state_kernel(const float* __restrict__ src,
                                  float* __restrict__ dst, int n) {
  int i = blockIdx.x * 256 + threadIdx.x;
  if (i < n) dst[i] = src[i];
}

__global__ void tr_w_kernel(const float* __restrict__ src, h16* __restrict__ dst,
                            int Kin, int Nout, int Kpad) {
  int i = blockIdx.x * 256 + threadIdx.x;
  if (i >= Nout * Kpad) return;
  int n = i / Kpad, kk = i - n * Kpad;
  dst[i] = (kk < Kin) ? (h16)src[(size_t)kk * Nout + n] : (h16)0.f;
}

// ---------------------------------------------------------------------------
// Encoder: one workgroup per (b, target node t). All 47 incoming edges (s,t)
// are processed as 3 resident M-tiles of 16 (s==t row gated to zero), so each
// weight B-fragment load is amortized over 3 WMMAs. Both MLP layers via WMMA,
// one-hot gate applied, reduced over edges in registers -> node_msg[b][t]
// (single write, no atomics, no zero-init pass).
// ---------------------------------------------------------------------------
__global__ void __launch_bounds__(256) enc_kernel(
    const float* __restrict__ state, float* __restrict__ node_msg,
    const float* __restrict__ etypes,
    const h16* __restrict__ W1t, const h16* __restrict__ W2t,
    const float* __restrict__ b1, const float* __restrict__ b2) {
  __shared__ __align__(16) h16 smsg[3][16][32];    // 3 edge_msg tiles
  __shared__ __align__(16) h16 sout1[3][16][HH];   // 3 layer-1 outputs (f16)
  __shared__ float sgate[2][NN];

  const int tid  = threadIdx.x;
  const int lane = tid & 31;
  const int wave = tid >> 5;
  const int b    = blockIdx.x / NN;
  const int t    = blockIdx.x % NN;

  // one-hot gates for all source nodes (edge id computed analytically)
  if (tid < 96) {
    int k = tid / NN, s = tid - k * NN;
    float g = 0.f;
    if (s != t) {
      int e = s * (NN - 1) + t - (t > s ? 1 : 0);
      g = etypes[((size_t)b * EE + e) * 3 + (k + 1)];
    }
    sgate[k][s] = g;
  }

  // build all 3 edge_msg tiles: row m -> [state[s], state[t]], s = mt*16+m
  for (int idx = tid; idx < 3 * 512; idx += 256) {
    int mt = idx >> 9, m = (idx >> 5) & 15, c = idx & 31;
    int s = mt * 16 + m;
    float v = (c < 16) ? state[(b * NN + s) * 16 + c]
                       : state[(b * NN + t) * 16 + (c - 16)];
    smsg[mt][m][c] = (h16)v;
  }
  __syncthreads();

  v16h a1[3];
#pragma unroll
  for (int mt = 0; mt < 3; ++mt) a1[mt] = ld_frag(&smsg[mt][0][0], 32);

  float colsum[2] = {0.f, 0.f};

  for (int k = 0; k < 2; ++k) {
    // layer 1: (16x32)@(32x256); one B fragment feeds 3 WMMAs
#pragma unroll
    for (int j = 0; j < 2; ++j) {
      int n = wave * 2 + j;
      v16h bf = ld_frag(W1t + (size_t)(k * HH + n * 16) * 32, 32);
      float bias = b1[k * HH + n * 16 + (lane & 15)];
#pragma unroll
      for (int mt = 0; mt < 3; ++mt) {
        v8f c1 = {};
        c1 = wmma_f16(a1[mt], bf, c1);
#pragma unroll
        for (int i = 0; i < 8; ++i) {
          int r = (lane < 16) ? i : (8 + i);
          float v = c1[i] + bias;
          sout1[mt][r][n * 16 + (lane & 15)] = (h16)(v > 0.f ? v : 0.f);
        }
      }
    }
    __syncthreads();

    // layer 2: (16x256)@(256x256); 8 K-steps, 6 live accumulators,
    // each B fragment amortized over 3 M-tiles
    v8f acc[2][3] = {};
    for (int kk = 0; kk < 8; ++kk) {
      v16h a2_0 = ld_frag(&sout1[0][0][kk * 32], HH);
      v16h a2_1 = ld_frag(&sout1[1][0][kk * 32], HH);
      v16h a2_2 = ld_frag(&sout1[2][0][kk * 32], HH);
#pragma unroll
      for (int j = 0; j < 2; ++j) {
        int n = wave * 2 + j;
        v16h bf = ld_frag(W2t + (size_t)k * HH * HH + (size_t)(n * 16) * HH + kk * 32, HH);
        acc[j][0] = wmma_f16(a2_0, bf, acc[j][0]);
        acc[j][1] = wmma_f16(a2_1, bf, acc[j][1]);
        acc[j][2] = wmma_f16(a2_2, bf, acc[j][2]);
      }
    }

    // relu + bias + one-hot gate, accumulate per-column sums over all edges
#pragma unroll
    for (int j = 0; j < 2; ++j) {
      int n = wave * 2 + j;
      float bias = b2[k * HH + n * 16 + (lane & 15)];
#pragma unroll
      for (int mt = 0; mt < 3; ++mt) {
#pragma unroll
        for (int i = 0; i < 8; ++i) {
          int r = (lane < 16) ? i : (8 + i);
          float v = acc[j][mt][i] + bias;
          v = v > 0.f ? v : 0.f;
          colsum[j] += v * sgate[k][mt * 16 + r];
        }
      }
    }
    __syncthreads();   // before next k rewrites sout1
  }

  // combine halves (lanes L / L+16 hold rows 0-7 / 8-15 of same column)
#pragma unroll
  for (int j = 0; j < 2; ++j) {
    float tot = colsum[j] + __shfl_xor(colsum[j], 16, 32);
    if (lane < 16)
      node_msg[(size_t)(b * NN + t) * HH + (wave * 2 + j) * 16 + lane] = tot;
  }
}

// ---------------------------------------------------------------------------
// Decoder: one workgroup per 16 rows of (B*N). Fuses concat -> d1 -> d2 ->
// out-projection -> state update -> output write.
// ---------------------------------------------------------------------------
__global__ void __launch_bounds__(256) dec_kernel(
    float* __restrict__ state, const float* __restrict__ node_msg,
    const h16* __restrict__ dW1t, const h16* __restrict__ dW2t,
    const h16* __restrict__ oWt,
    const float* __restrict__ db1, const float* __restrict__ db2,
    const float* __restrict__ ob, float* __restrict__ out, int t) {
  __shared__ __align__(16) h16 sin_[16][288];   // node_in, K padded 272->288
  __shared__ __align__(16) h16 sd1[16][HH];
  __shared__ __align__(16) h16 sd2[16][HH];
  __shared__ float spart[8][16][16];            // per-wave partials, det. reduce

  const int tid  = threadIdx.x;
  const int lane = tid & 31;
  const int wave = tid >> 5;
  const int row0 = blockIdx.x * 16;

  for (int idx = tid; idx < 16 * 288; idx += 256) {
    int m = idx / 288, c = idx - m * 288;
    int gr = row0 + m;
    float v;
    if (c < 16)        v = state[gr * 16 + c];
    else if (c < 272)  v = node_msg[(size_t)gr * HH + (c - 16)];
    else               v = 0.f;
    sin_[m][c] = (h16)v;
  }
  __syncthreads();

  { // d1 = relu(node_in @ dec_W1 + b1), K=288 (9 steps; pad cols are zero)
    v8f acc[2] = {};
    for (int kk = 0; kk < 9; ++kk) {
      v16h a = ld_frag(&sin_[0][kk * 32], 288);
#pragma unroll
      for (int j = 0; j < 2; ++j) {
        int n = wave * 2 + j;
        v16h bf = ld_frag(dW1t + (size_t)(n * 16) * 288 + kk * 32, 288);
        acc[j] = wmma_f16(a, bf, acc[j]);
      }
    }
#pragma unroll
    for (int j = 0; j < 2; ++j) {
      int n = wave * 2 + j;
      float bias = db1[n * 16 + (lane & 15)];
#pragma unroll
      for (int i = 0; i < 8; ++i) {
        int r = (lane < 16) ? i : (8 + i);
        float v = acc[j][i] + bias;
        sd1[r][n * 16 + (lane & 15)] = (h16)(v > 0.f ? v : 0.f);
      }
    }
  }
  __syncthreads();

  { // d2 = relu(d1 @ dec_W2 + b2), K=256
    v8f acc[2] = {};
    for (int kk = 0; kk < 8; ++kk) {
      v16h a = ld_frag(&sd1[0][kk * 32], HH);
#pragma unroll
      for (int j = 0; j < 2; ++j) {
        int n = wave * 2 + j;
        v16h bf = ld_frag(dW2t + (size_t)(n * 16) * HH + kk * 32, HH);
        acc[j] = wmma_f16(a, bf, acc[j]);
      }
    }
#pragma unroll
    for (int j = 0; j < 2; ++j) {
      int n = wave * 2 + j;
      float bias = db2[n * 16 + (lane & 15)];
#pragma unroll
      for (int i = 0; i < 8; ++i) {
        int r = (lane < 16) ? i : (8 + i);
        float v = acc[j][i] + bias;
        sd2[r][n * 16 + (lane & 15)] = (h16)(v > 0.f ? v : 0.f);
      }
    }
  }
  __syncthreads();

  { // out = d2 @ out_W : each wave one K-slice of 32, deterministic LDS reduce
    v16h a  = ld_frag(&sd2[0][wave * 32], HH);
    v16h bf = ld_frag(oWt + wave * 32, HH);
    v8f c = {};
    c = wmma_f16(a, bf, c);
#pragma unroll
    for (int i = 0; i < 8; ++i) {
      int r = (lane < 16) ? i : (8 + i);
      spart[wave][r][lane & 15] = c[i];
    }
  }
  __syncthreads();

  { // state += proj + out_b ; also write output slot t
    int m = tid >> 4, d = tid & 15;
    int gr = row0 + m;
    int b = gr / NN, n = gr - b * NN;
    float s = 0.f;
#pragma unroll
    for (int w = 0; w < 8; ++w) s += spart[w][m][d];
    float nv = state[gr * 16 + d] + s + ob[d];
    state[gr * 16 + d] = nv;
    out[(((size_t)b * PRED + t) * NN + n) * 16 + d] = nv;
  }
}

// ---------------------------------------------------------------------------
extern "C" void kernel_launch(void* const* d_in, const int* in_sizes, int n_in,
                              void* d_out, int out_size, void* d_ws, size_t ws_size,
                              hipStream_t stream) {
  (void)in_sizes; (void)n_in; (void)out_size; (void)ws_size;
  const float* time_segs  = (const float*)d_in[0];
  const float* edge_types = (const float*)d_in[1];
  // d_in[2] src_idx, d_in[3] tgt_idx: edge ids derived analytically
  const float* enc_W1 = (const float*)d_in[4];
  const float* enc_b1 = (const float*)d_in[5];
  const float* enc_W2 = (const float*)d_in[6];
  const float* enc_b2 = (const float*)d_in[7];
  const float* dec_W1 = (const float*)d_in[8];
  const float* dec_b1 = (const float*)d_in[9];
  const float* dec_W2 = (const float*)d_in[10];
  const float* dec_b2 = (const float*)d_in[11];
  const float* out_W  = (const float*)d_in[12];
  const float* out_b  = (const float*)d_in[13];
  float* out = (float*)d_out;

  char* ws = (char*)d_ws;
  float* state    = (float*)(ws + 0);        //  98304 B
  float* node_msg = (float*)(ws + 98304);    // 1572864 B
  h16* W1t  = (h16*)(ws + 1671168);          //  32768 B  (2 x 256 x 32)
  h16* W2t  = (h16*)(ws + 1703936);          // 262144 B  (2 x 256 x 256)
  h16* dW1t = (h16*)(ws + 1966080);          // 147456 B  (256 x 288)
  h16* dW2t = (h16*)(ws + 2113536);          // 131072 B  (256 x 256)
  h16* oWt  = (h16*)(ws + 2244608);          //   8192 B  (16 x 256)

  copy_state_kernel<<<96, 256, 0, stream>>>(time_segs, state, BB * NN * 16);
  for (int k = 0; k < 2; ++k) {
    tr_w_kernel<<<(256 * 32 + 255) / 256, 256, 0, stream>>>(
        enc_W1 + (size_t)k * 32 * 256, W1t + (size_t)k * 256 * 32, 32, 256, 32);
    tr_w_kernel<<<(256 * 256 + 255) / 256, 256, 0, stream>>>(
        enc_W2 + (size_t)k * 256 * 256, W2t + (size_t)k * 256 * 256, 256, 256, 256);
  }
  tr_w_kernel<<<(256 * 288 + 255) / 256, 256, 0, stream>>>(dec_W1, dW1t, 272, 256, 288);
  tr_w_kernel<<<(256 * 256 + 255) / 256, 256, 0, stream>>>(dec_W2, dW2t, 256, 256, 256);
  tr_w_kernel<<<(16 * 256 + 255) / 256, 256, 0, stream>>>(out_W, oWt, 256, 16, 256);

  for (int t = 0; t < PRED; ++t) {
    enc_kernel<<<BB * NN, 256, 0, stream>>>(state, node_msg, edge_types,
                                            W1t, W2t, enc_b1, enc_b2);
    dec_kernel<<<BB * NN / 16, 256, 0, stream>>>(state, node_msg, dW1t, dW2t, oWt,
                                                 dec_b1, dec_b2, out_b, out, t);
  }
}
